// IntentDetection_80788334838032
// MI455X (gfx1250) — matrix operand order repaired and checked by reading, not compiled, not run on hardware
//
#include <hip/hip_runtime.h>
#include <hip/hip_bf16.h>

typedef __attribute__((ext_vector_type(16))) __bf16 v16bf;
typedef __attribute__((ext_vector_type(8)))  float  v8f;

#define B_   128
#define T_   512
#define E_   256
#define H_   256
#define G4H  1024   // 4*H
#define C_   32

union AF { v16bf v; uint4 u[2]; };

// ---------------------------------------------------------------- kernel 1
// Embedding gather + cast to bf16.  x_bf16[b*T + t][e] = emb[tok][e]
__global__ void k_embed(const int* __restrict__ tokens,
                        const float* __restrict__ emb,
                        __bf16* __restrict__ xb) {
    const int bt = blockIdx.x;          // 0 .. B*T-1
    const int e  = threadIdx.x;         // 0 .. 255
    const int tok = tokens[bt];
    xb[(size_t)bt * E_ + e] = (__bf16)emb[(size_t)tok * E_ + e];
}

// ---------------------------------------------------------------- kernel 2
// Transpose + cast weights to bf16 column-major: WT[n][k] = W[k][n]
__global__ void k_prep(const float* __restrict__ Wf, const float* __restrict__ Wb,
                       const float* __restrict__ Uf, const float* __restrict__ Ub,
                       __bf16* __restrict__ WTf, __bf16* __restrict__ WTb,
                       __bf16* __restrict__ UTf, __bf16* __restrict__ UTb) {
    const int n = blockIdx.x;           // 0 .. 1023
    const int k = threadIdx.x;          // 0 .. 255
    WTf[(size_t)n * E_ + k] = (__bf16)Wf[(size_t)k * G4H + n];
    WTb[(size_t)n * E_ + k] = (__bf16)Wb[(size_t)k * G4H + n];
    UTf[(size_t)n * H_ + k] = (__bf16)Uf[(size_t)k * G4H + n];
    UTb[(size_t)n * H_ + k] = (__bf16)Ub[(size_t)k * G4H + n];
}

// ---------------------------------------------------------------- kernel 3
// xw[dir][t][b][n] = x[b][tsrc] @ W + bias  (tsrc reversed for backward dir)
// grid (8, 8, 1024): x = n-block (128 cols), y = b-tile (16 rows), z = dir*T + t
// block 256 = 8 waves; wave w handles cols [128*bx + 16*w, +16), K = 256.
__global__ void k_xproj(const __bf16* __restrict__ xb,
                        const __bf16* __restrict__ WTf, const __bf16* __restrict__ WTb,
                        const float* __restrict__ bf_, const float* __restrict__ bb_,
                        float* __restrict__ xwf, float* __restrict__ xwb) {
    const int wave = threadIdx.x >> 5;
    const int lane = threadIdx.x & 31;
    const int r    = lane & 15;
    const int hi   = lane >> 4;

    const int zi  = blockIdx.z;
    const int dir = zi >> 9;            // 0 fwd, 1 bwd
    const int t   = zi & 511;
    const int b0  = blockIdx.y * 16;
    const int n0  = blockIdx.x * 128 + wave * 16;

    const __bf16* WT   = dir ? WTb : WTf;
    const float*  bias = dir ? bb_ : bf_;
    float*        xw   = dir ? xwb : xwf;
    const int     tsrc = dir ? (T_ - 1 - t) : t;

    const __bf16* arow = xb + ((size_t)(b0 + r) * T_ + tsrc) * E_;
    const __bf16* brow = WT + (size_t)(n0 + r) * E_;

    v8f acc = {0.f, 0.f, 0.f, 0.f, 0.f, 0.f, 0.f, 0.f};
#pragma unroll
    for (int kt = 0; kt < 8; ++kt) {
        AF a, b;
        const int ka0 = kt * 32 + hi * 8;           // ISA 16-bit A layout
        a.u[0] = *(const uint4*)(arow + ka0);
        a.u[1] = *(const uint4*)(arow + ka0 + 16);
        const int kb = kt * 32 + hi * 8;            // ISA 16-bit B layout
        b.u[0] = *(const uint4*)(brow + kb);
        b.u[1] = *(const uint4*)(brow + kb + 16);
        acc = __builtin_amdgcn_wmma_f32_16x16x32_bf16(
                  false, a.v, false, b.v, (short)0, acc, false, false);
    }
    const float bv   = bias[n0 + r];
    float*      orow = xw + ((size_t)t * B_ + b0) * G4H + (n0 + r);
#pragma unroll
    for (int v = 0; v < 8; ++v) {                    // C layout: M = v + 8*hi, N = r
        orow[(size_t)(v + 8 * hi) * G4H] = acc[v] + bv;
    }
}

// ---------------------------------------------------------------- kernel 4
// Persistent recurrence. grid 16 blocks: blockIdx.x = dir*8 + batch_tile.
// Per step: z = xw[t] + h@U (WMMA), then fp32 gate math in LDS.
__global__ void k_lstm(const float* __restrict__ xwf, const float* __restrict__ xwb,
                       const __bf16* __restrict__ UTf, const __bf16* __restrict__ UTb,
                       float* __restrict__ hcat) {
    __shared__ __align__(16) float  zbuf[16][G4H];   // 64 KB
    __shared__ __align__(16) float  cbuf[16][H_];    // 16 KB
    __shared__ __align__(16) __bf16 hbuf[16][H_];    //  8 KB

    const int tid  = threadIdx.x;
    const int wave = tid >> 5;
    const int lane = tid & 31;
    const int r    = lane & 15;
    const int hi   = lane >> 4;

    const int dir = blockIdx.x >> 3;
    const int b0  = (blockIdx.x & 7) * 16;
    const float*  xw = dir ? xwb : xwf;
    const __bf16* UT = dir ? UTb : UTf;

    for (int i = tid; i < 16 * H_; i += 256) {
        (&cbuf[0][0])[i] = 0.f;
        (&hbuf[0][0])[i] = (__bf16)0.f;
    }
    __syncthreads();

    const int em  = tid >> 4;            // elementwise row 0..15
    const int ec0 = (tid & 15) * 16;     // elementwise col base

    for (int t = 0; t < T_; ++t) {
        // ---- A fragments of h (shared across all 8 column tiles) ----
        AF a[8];
        const __bf16* arow = &hbuf[r][0];
#pragma unroll
        for (int kt = 0; kt < 8; ++kt) {
            const int ka0 = kt * 32 + hi * 8;
            a[kt].u[0] = *(const uint4*)(arow + ka0);
            a[kt].u[1] = *(const uint4*)(arow + ka0 + 16);
        }
        const float* xrow0 = xw + ((size_t)t * B_ + b0) * G4H;
#pragma unroll
        for (int ct = 0; ct < 8; ++ct) {
            const int n0 = wave * 128 + ct * 16;
            // init accumulator from xw[t] (already includes x@W + bias)
            const float* xcol = xrow0 + n0 + r;
            v8f acc;
#pragma unroll
            for (int v = 0; v < 8; ++v) acc[v] = xcol[(size_t)(v + 8 * hi) * G4H];

            const __bf16* brow = UT + (size_t)(n0 + r) * H_;
#pragma unroll
            for (int kt = 0; kt < 8; ++kt) {
                AF b;
                const int kb = kt * 32 + hi * 8;
                b.u[0] = *(const uint4*)(brow + kb);
                b.u[1] = *(const uint4*)(brow + kb + 16);
                acc = __builtin_amdgcn_wmma_f32_16x16x32_bf16(
                          false, a[kt].v, false, b.v, (short)0, acc, false, false);
            }
#pragma unroll
            for (int v = 0; v < 8; ++v) zbuf[v + 8 * hi][n0 + r] = acc[v];
        }
        __syncthreads();

        // ---- gate math: thread owns row em, cols [ec0, ec0+16) ----
        for (int j = 0; j < 16; ++j) {
            const int hc = ec0 + j;
            const float zi_ = zbuf[em][hc];
            const float zf_ = zbuf[em][H_ + hc];
            const float zg_ = zbuf[em][2 * H_ + hc];
            const float zo_ = zbuf[em][3 * H_ + hc];
            const float ig = 1.f / (1.f + __expf(-zi_));
            const float fg = 1.f / (1.f + __expf(-zf_));
            const float gg = tanhf(zg_);
            const float og = 1.f / (1.f + __expf(-zo_));
            const float c  = fg * cbuf[em][hc] + ig * gg;
            const float h  = og * tanhf(c);
            cbuf[em][hc] = c;
            hbuf[em][hc] = (__bf16)h;
            if (t == T_ - 1)
                hcat[(size_t)(b0 + em) * (2 * H_) + dir * H_ + hc] = h;
        }
        __syncthreads();
    }
}

// ---------------------------------------------------------------- kernel 5
// d1 = relu(hcat @ W1 + b1); out = softmax(d1 @ W2 + b2). One block per batch row.
__global__ void k_head(const float* __restrict__ hcat,
                       const float* __restrict__ W1, const float* __restrict__ b1,
                       const float* __restrict__ W2, const float* __restrict__ b2,
                       float* __restrict__ out) {
    const int b = blockIdx.x;
    const int j = threadIdx.x;               // 0..255
    __shared__ float hrow[2 * H_];
    __shared__ float d1[H_];
    __shared__ float logits[C_];
    __shared__ float ex[C_];

    hrow[j]       = hcat[(size_t)b * 2 * H_ + j];
    hrow[j + 256] = hcat[(size_t)b * 2 * H_ + j + 256];
    __syncthreads();

    float acc = b1[j];
    for (int k = 0; k < 2 * H_; ++k) acc += hrow[k] * W1[(size_t)k * H_ + j];
    d1[j] = fmaxf(acc, 0.f);
    __syncthreads();

    if (j < C_) {
        float l = b2[j];
        for (int k = 0; k < H_; ++k) l += d1[k] * W2[(size_t)k * C_ + j];
        logits[j] = l;
    }
    __syncthreads();

    if (j == 0) {
        float mx = logits[0];
        for (int c = 1; c < C_; ++c) mx = fmaxf(mx, logits[c]);
        float s = 0.f;
        for (int c = 0; c < C_; ++c) { float e = __expf(logits[c] - mx); ex[c] = e; s += e; }
        const float inv = 1.f / s;
        for (int c = 0; c < C_; ++c) out[(size_t)b * C_ + c] = ex[c] * inv;
    }
}

// ---------------------------------------------------------------- launcher
extern "C" void kernel_launch(void* const* d_in, const int* in_sizes, int n_in,
                              void* d_out, int out_size, void* d_ws, size_t ws_size,
                              hipStream_t stream) {
    (void)in_sizes; (void)n_in; (void)out_size; (void)ws_size;
    const int*   tokens = (const int*)  d_in[0];
    const float* emb    = (const float*)d_in[1];
    const float* Wf     = (const float*)d_in[2];
    const float* Uf     = (const float*)d_in[3];
    const float* bf_    = (const float*)d_in[4];
    const float* Wb     = (const float*)d_in[5];
    const float* Ub     = (const float*)d_in[6];
    const float* bb_    = (const float*)d_in[7];
    const float* W1     = (const float*)d_in[8];
    const float* b1     = (const float*)d_in[9];
    const float* W2     = (const float*)d_in[10];
    const float* b2     = (const float*)d_in[11];
    float* out = (float*)d_out;

    char*  ws  = (char*)d_ws;
    size_t off = 0;
    auto carve = [&](size_t bytes) -> void* {
        void* p = ws + off;
        off = (off + bytes + 255) & ~(size_t)255;
        return p;
    };
    __bf16* xb   = (__bf16*)carve((size_t)B_ * T_ * E_ * 2);        //  32 MB
    __bf16* WTf  = (__bf16*)carve((size_t)G4H * E_ * 2);
    __bf16* WTb  = (__bf16*)carve((size_t)G4H * E_ * 2);
    __bf16* UTf  = (__bf16*)carve((size_t)G4H * H_ * 2);
    __bf16* UTb  = (__bf16*)carve((size_t)G4H * H_ * 2);
    float*  xwf  = (float*) carve((size_t)T_ * B_ * G4H * 4);       // 256 MB
    float*  xwb  = (float*) carve((size_t)T_ * B_ * G4H * 4);       // 256 MB
    float*  hcat = (float*) carve((size_t)B_ * 2 * H_ * 4);

    k_embed<<<B_ * T_, 256, 0, stream>>>(tokens, emb, xb);
    k_prep <<<G4H, 256, 0, stream>>>(Wf, Wb, Uf, Ub, WTf, WTb, UTf, UTb);
    k_xproj<<<dim3(8, 8, 2 * T_), 256, 0, stream>>>(xb, WTf, WTb, bf_, bb_, xwf, xwb);
    k_lstm <<<16, 256, 0, stream>>>(xwf, xwb, UTf, UTb, hcat);
    k_head <<<B_, 256, 0, stream>>>(hcat, W1, b1, W2, b2, out);
}
